// HyberNet_88734024335775
// MI455X (gfx1250) — compile-verified, matrix operand order checked by reading, not run on hardware
//
#include <hip/hip_runtime.h>

// ---------------------------------------------------------------------------
// CDNA5 / gfx1250 HyberNet implementation.
// - All matmul-like work via v_wmma_f32_16x16x32_f16 (wave32, fp32 accum).
// - Weights pre-cast to f16 in HBM; A-panel staged into LDS with
//   global_load_async_to_lds_b128 (ASYNCcnt), overlapped with im2col B staging.
// ---------------------------------------------------------------------------

typedef _Float16 v16h __attribute__((ext_vector_type(16)));
typedef _Float16 h8   __attribute__((ext_vector_type(8)));
typedef float    v8f  __attribute__((ext_vector_type(8)));

#define BM  64      // block M (Co)
#define BND 128     // block N (spatial)
#define BK  32      // block K
#define LDK (BK + 8)   // padded row length (16B-aligned: 40*2 = 80 bytes)
#define GROUPS 8

// ---------------------------------------------------------------------------
// Async copy of 16 bytes global -> LDS (gfx1250 GLOBAL_LOAD_ASYNC_TO_LDS_B128,
// tracked by ASYNCcnt). vdst holds the LDS byte address (flat shared-aperture
// address truncated to 32 bits per the ISA flat->LDS mapping).
// ---------------------------------------------------------------------------
__device__ __forceinline__ void async_copy_b128(const void* gptr, void* lptr)
{
    unsigned laddr = (unsigned)(unsigned long long)lptr;
    unsigned long long ga = (unsigned long long)gptr;
    asm volatile("global_load_async_to_lds_b128 %0, %1, off"
                 :: "v"(laddr), "v"(ga)
                 : "memory");
}

__device__ __forceinline__ void wait_async0()
{
    asm volatile("s_wait_asynccnt 0x0" ::: "memory");
}

// ---------------------------------------------------------------------------
// Load one 16x16x32 f16 fragment row from LDS.
// Per the ISA 16-bit A/B layout (7.12.2), lane (half = lane>>4) needs the two
// contiguous 8-element runs K in [8*half, 8*half+8) and [16+8*half, 24+8*half)
// of its row -> two ds_load_b128 + a register shuffle.
// ---------------------------------------------------------------------------
__device__ __forceinline__ v16h load_frag(const _Float16* __restrict__ row,
                                          int half)
{
    const h8* p = (const h8*)row;
    h8 lo = p[half];
    h8 hi = p[half + 2];
    return __builtin_shufflevector(lo, hi, 0, 1, 2, 3, 4, 5, 6, 7,
                                   8, 9, 10, 11, 12, 13, 14, 15);
}

// fp32 -> f16 tensor cast (used once per weight tensor per launch)
__global__ __launch_bounds__(256)
void k_cast_f16(const float* __restrict__ X, _Float16* __restrict__ Y,
                long long n)
{
    long long i = blockIdx.x * 256ll + threadIdx.x;
    if (i < n) Y[i] = (_Float16)X[i];
}

// ---------------------------------------------------------------------------
// Implicit-GEMM convolution, f16 WMMA, fp32 accumulate.
//   Y[b, co, p] = sum_{ci,r,s} W[co, ci, r, s] * X[b, ci, ...im2col...]
// GEMM view: M = Co, N = Hout*Wout, K = Ci*kh*kw.
// 256 threads = 8 waves tile the 64x128 output as 4x8 16x16 WMMA tiles; each
// wave owns one M-tile and four N-tiles (4 WMMA / K-step off one A fragment).
// A (f16 weights): 64 rows x 64B -> one async b128 per thread per K-step,
// overlapped with the synchronous im2col B staging (f32->f16 on the fly).
// B staged transposed [n][k] so fragment loads are two 128-bit DS reads.
// Also used as a batched GEMM (kh=kw=1) for the attention qkv projection.
// ---------------------------------------------------------------------------
__global__ __launch_bounds__(256)
void k_conv_wmma(const float* __restrict__ X, const _Float16* __restrict__ W16,
                 const float* __restrict__ bias, float* __restrict__ Y,
                 int Ci, int Hin, int Win, int Co, int Hout, int Wout,
                 int kh, int kw, int stride, int pad,
                 long long bsx, long long bsy)
{
    const int K   = Ci * kh * kw;
    const int NSP = Hout * Wout;
    const int n0  = blockIdx.x * BND;
    const int m0  = blockIdx.y * BM;
    const int b   = blockIdx.z;
    X += (long long)b * bsx;
    Y += (long long)b * bsy;

    __shared__ alignas(16) _Float16 As[BM][LDK];    // [m][k]
    __shared__ alignas(16) _Float16 Bs[BND][LDK];   // [n][k] (transposed)

    const int tid  = threadIdx.x;
    const int wave = tid >> 5;
    const int lane = tid & 31;
    const int l16  = lane & 15;
    const int half = lane >> 4;

    const int mbase = (wave & 3) * 16;    // M tile within block
    const int nbase = (wave >> 2) * 64;   // four N tiles within block

    const v8f vzero = {0.f, 0.f, 0.f, 0.f, 0.f, 0.f, 0.f, 0.f};
    v8f acc[4] = {vzero, vzero, vzero, vzero};

    const int  khw    = kh * kw;
    const bool cleanK = (K & 31) == 0;    // true for every conv except the stem

    // fixed per-thread A-chunk: 256 threads cover 64 rows x 4 16B chunks
    const int am = tid >> 2;              // row 0..63
    const int ak = (tid & 3) * 8;         // k offset 0,8,16,24
    const int agm = m0 + am;

    for (int k0 = 0; k0 < K; k0 += BK) {
        // ---- A panel: async b128 copy (weights already f16 in HBM) ----
        if (cleanK && agm < Co) {
            async_copy_b128(W16 + (long long)agm * K + (k0 + ak), &As[am][ak]);
        } else {
#pragma unroll
            for (int u = 0; u < 8; ++u) {
                int gk = k0 + ak + u;
                _Float16 v = (_Float16)0.f;
                if (agm < Co && gk < K) v = W16[(long long)agm * K + gk];
                As[am][ak + u] = v;
            }
        }

        // ---- B panel: im2col gather, fp32 -> f16, store transposed [n][k] ----
        for (int i = tid; i < BK * BND; i += 256) {
            int k = i >> 7, n = i & 127;
            int gk = k0 + k, gn = n0 + n;
            float v = 0.f;
            if (gk < K && gn < NSP) {
                int ci  = gk / khw;
                int rem = gk - ci * khw;
                int r   = rem / kw;
                int s   = rem - r * kw;
                int oh  = gn / Wout;
                int ow  = gn - oh * Wout;
                int ih  = oh * stride - pad + r;
                int iw  = ow * stride - pad + s;
                if ((unsigned)ih < (unsigned)Hin && (unsigned)iw < (unsigned)Win)
                    v = X[((long long)ci * Hin + ih) * Win + iw];
            }
            Bs[n][k] = (_Float16)v;
        }

        wait_async0();          // A panel landed in LDS
        __syncthreads();

        v16h a = load_frag(&As[mbase + l16][0], half);
#pragma unroll
        for (int t = 0; t < 4; ++t) {
            v16h bf = load_frag(&Bs[nbase + t * 16 + l16][0], half);
            acc[t] = __builtin_amdgcn_wmma_f32_16x16x32_f16(
                false, a, false, bf, (short)0, acc[t], false, false);
        }
        __syncthreads();
    }

    // Epilogue: C/D layout -> lane = N%16 (+16 for M>=8), VGPR r -> M = r (+8)
#pragma unroll
    for (int t = 0; t < 4; ++t) {
        int ncol = n0 + nbase + t * 16 + l16;
        if (ncol < NSP) {
#pragma unroll
            for (int v2 = 0; v2 < 8; ++v2) {
                int m = m0 + mbase + half * 8 + v2;
                if (m < Co) {
                    float o = acc[t][v2];
                    if (bias) o += bias[m];
                    Y[(long long)m * NSP + ncol] = o;
                }
            }
        }
    }
}

// ---------------------------------------------------------------------------
// Block reduction helper (256 or 128 threads).
// ---------------------------------------------------------------------------
__device__ __forceinline__ float block_reduce_sum(float v, float* sm)
{
    int tid = threadIdx.x;
    sm[tid] = v;
    __syncthreads();
    for (int s = blockDim.x >> 1; s > 0; s >>= 1) {
        if (tid < s) sm[tid] += sm[tid + s];
        __syncthreads();
    }
    float r = sm[0];
    __syncthreads();
    return r;
}

// ---------------------------------------------------------------------------
// Batch-norm statistics: X viewed as (outer, C, inner), channel c at
// o*ostride + c*inner. Produces fused scale/shift so the apply pass is a FMA.
// ---------------------------------------------------------------------------
__global__ __launch_bounds__(256)
void k_bn_stats(const float* __restrict__ X, const float* __restrict__ gam,
                const float* __restrict__ bet, float* __restrict__ scale,
                float* __restrict__ shift, int inner, int outer, long long ostride)
{
    int c = blockIdx.x;
    float s = 0.f, s2 = 0.f;
    for (int o = 0; o < outer; ++o) {
        const float* p = X + (long long)o * ostride + (long long)c * inner;
        for (int i = threadIdx.x; i < inner; i += 256) {
            float v = p[i];
            s += v; s2 += v * v;
        }
    }
    __shared__ float sm[256];
    s  = block_reduce_sum(s, sm);
    s2 = block_reduce_sum(s2, sm);
    if (threadIdx.x == 0) {
        float n = (float)inner * (float)outer;
        float mean = s / n;
        float var  = s2 / n - mean * mean;
        float sc   = gam[c] * rsqrtf(var + 1e-5f);
        scale[c] = sc;
        shift[c] = bet[c] - mean * sc;
    }
}

// ---------------------------------------------------------------------------
// Fused elementwise: y = relu?( bn?(x) + res? ). sc==null -> identity BN.
// X layout (outer, C, inner) contiguous.
// ---------------------------------------------------------------------------
__global__ __launch_bounds__(256)
void k_bn_apply(const float* __restrict__ X, const float* __restrict__ sc,
                const float* __restrict__ sh, const float* __restrict__ res,
                float* __restrict__ Y, long long total, int C, long long inner,
                int relu)
{
    long long idx = blockIdx.x * 256ll + threadIdx.x;
    if (idx >= total) return;
    float v = X[idx];
    if (sc) {
        int c = (int)((idx / inner) % C);
        v = v * sc[c] + sh[c];
    }
    if (res) v += res[idx];
    if (relu) v = fmaxf(v, 0.f);
    Y[idx] = v;
}

// ---------------------------------------------------------------------------
// Axial gather: X (C,H,W) -> T (B,C,L).
// width=0: b=w, t=h (height attention). width=1: b=h, t=w (width attention).
// ---------------------------------------------------------------------------
__global__ __launch_bounds__(256)
void k_axial_gather(const float* __restrict__ X, float* __restrict__ T,
                    int C, int H, int W, int width)
{
    long long total = (long long)C * H * W;
    long long idx = blockIdx.x * 256ll + threadIdx.x;
    if (idx >= total) return;
    int L = width ? W : H;
    int t = (int)(idx % L);
    int c = (int)((idx / L) % C);
    int b = (int)(idx / ((long long)L * C));
    int h = width ? b : t;
    int w = width ? t : b;
    T[idx] = X[((long long)c * H + h) * W + w];
}

// ---------------------------------------------------------------------------
// Axial scores for one (b,g,i,j): qk, qr, kr' from BN'd qkv + rel embedding.
//   qk = sum_c q[c,i] k[c,j]
//   qr = sum_c q[c,i] rel[c, i-j+L-1]
//   kr'= sum_c k[c,j] rel[hgp+c, j-i+L-1]   (kr after swapaxes(2,3))
// ---------------------------------------------------------------------------
__device__ __forceinline__ void axial_scores(const float* __restrict__ qg,
                                             const float* __restrict__ kg,
                                             const float* __restrict__ rel,
                                             int hgp, int L, int i, int j,
                                             float& aqk, float& aqr, float& akr)
{
    int R  = 2 * L - 1;
    int dq = i - j + L - 1;
    int dk = j - i + L - 1;
    float s0 = 0.f, s1 = 0.f, s2 = 0.f;
    for (int c = 0; c < hgp; ++c) {
        float qi = qg[c * L + i];
        float kj = kg[c * L + j];
        s0 += qi * kj;
        s1 += qi * rel[c * R + dq];
        s2 += kj * rel[(hgp + c) * R + dk];
    }
    aqk = s0; aqr = s1; akr = s2;
}

// BN stats over the 3G=24 stacked score channels, recomputing scores on the
// fly (avoids materializing the (B,24,L,L) tensor: ~158 MB at level 1).
__global__ __launch_bounds__(256)
void k_score_stats(const float* __restrict__ QKV, const float* __restrict__ rel,
                   const float* __restrict__ gam, const float* __restrict__ bet,
                   float* __restrict__ scale, float* __restrict__ shift,
                   int C, int L, int B, int gp)
{
    int ch = blockIdx.x;            // 0..23
    int p = ch >> 3, g = ch & 7;
    int hgp = gp >> 1;
    long long per = (long long)L * L;
    long long tot = (long long)B * per;
    float s = 0.f, s2 = 0.f;
    for (long long idx = threadIdx.x; idx < tot; idx += 256) {
        int j = (int)(idx % L);
        int i = (int)((idx / L) % L);
        int b = (int)(idx / per);
        const float* qg = QKV + ((long long)b * 2 * C + (long long)g * 2 * gp) * L;
        float aqk, aqr, akr;
        axial_scores(qg, qg + (long long)hgp * L, rel, hgp, L, i, j, aqk, aqr, akr);
        float v = (p == 0) ? aqk : ((p == 1) ? aqr : akr);
        s += v; s2 += v * v;
    }
    __shared__ float sm[256];
    s  = block_reduce_sum(s, sm);
    s2 = block_reduce_sum(s2, sm);
    if (threadIdx.x == 0) {
        float n = (float)tot;
        float mean = s / n;
        float var  = s2 / n - mean * mean;
        float sc   = gam[ch] * rsqrtf(var + 1e-5f);
        scale[ch] = sc;
        shift[ch] = bet[ch] - mean * sc;
    }
}

// Fused: recompute scores, apply BN, sum the 3 parts, row softmax -> SIM.
// One 128-thread block per (b,g,i) row; L <= 118.
__global__ __launch_bounds__(128)
void k_softmax_scores(const float* __restrict__ QKV, const float* __restrict__ rel,
                      const float* __restrict__ sc, const float* __restrict__ sh,
                      float* __restrict__ SIM, int C, int L, int gp)
{
    int bid = blockIdx.x;           // b*8*L + g*L + i
    int i = bid % L;
    int t = bid / L;
    int g = t & 7;
    int b = t >> 3;
    int hgp = gp >> 1;
    const float* qg = QKV + ((long long)b * 2 * C + (long long)g * 2 * gp) * L;
    const float* kg = qg + (long long)hgp * L;

    __shared__ float buf[128];
    __shared__ float red[128];
    int j = threadIdx.x;
    float z = -3.0e38f;
    if (j < L) {
        float aqk, aqr, akr;
        axial_scores(qg, kg, rel, hgp, L, i, j, aqk, aqr, akr);
        z = aqk * sc[g] + sh[g] + aqr * sc[8 + g] + sh[8 + g]
          + akr * sc[16 + g] + sh[16 + g];
        buf[j] = z;
    }
    red[threadIdx.x] = z;
    __syncthreads();
    for (int s = 64; s > 0; s >>= 1) {
        if (threadIdx.x < s)
            red[threadIdx.x] = fmaxf(red[threadIdx.x], red[threadIdx.x + s]);
        __syncthreads();
    }
    float m = red[0];
    __syncthreads();
    float e = 0.f;
    if (j < L) { e = __expf(buf[j] - m); buf[j] = e; }
    red[threadIdx.x] = e;
    __syncthreads();
    for (int s = 64; s > 0; s >>= 1) {
        if (threadIdx.x < s) red[threadIdx.x] += red[threadIdx.x + s];
        __syncthreads();
    }
    float inv = 1.f / red[0];
    if (j < L) SIM[(long long)bid * L + j] = buf[j] * inv;
}

// sv/sve: O[b, (g*gp+c)*2 + {0,1}, i] = sum_j SIM[b,g,i,j] * {v[c,j], ve[c,i,j]}
__global__ __launch_bounds__(256)
void k_attn_apply(const float* __restrict__ SIM, const float* __restrict__ QKV,
                  const float* __restrict__ rel, float* __restrict__ O,
                  int C, int L, int gp, long long total)
{
    long long idx = blockIdx.x * 256ll + threadIdx.x;
    if (idx >= total) return;
    int i = (int)(idx % L);
    int c = (int)((idx / L) % gp);
    int g = (int)((idx / ((long long)L * gp)) & 7);
    int b = (int)(idx / ((long long)L * gp * 8));
    int R = 2 * L - 1;
    const float* v    = QKV + ((long long)b * 2 * C + (long long)g * 2 * gp + gp) * L;
    const float* srow = SIM + (((long long)(b * 8 + g)) * L + i) * L;
    const float* ve   = rel + (long long)(gp + c) * R + (i + L - 1);
    float asv = 0.f, asve = 0.f;
    for (int j = 0; j < L; ++j) {
        float s = srow[j];
        asv  += s * v[c * L + j];
        asve += s * ve[-j];
    }
    long long och = (long long)b * 2 * C + (long long)(g * gp + c) * 2;
    O[och * L + i]       = asv;
    O[(och + 1) * L + i] = asve;
}

// BN-apply + pair-sum + transpose back: Y[c,h,w] = sum_s bn(O[b, 2c+s, t])
__global__ __launch_bounds__(256)
void k_axial_out(const float* __restrict__ O, const float* __restrict__ sc,
                 const float* __restrict__ sh, float* __restrict__ Y,
                 int C, int H, int W, int width)
{
    long long total = (long long)C * H * W;
    long long idx = blockIdx.x * 256ll + threadIdx.x;
    if (idx >= total) return;
    int w = (int)(idx % W);
    int h = (int)((idx / W) % H);
    int c = (int)(idx / ((long long)W * H));
    int L = width ? W : H;
    int b = width ? h : w;
    int t = width ? w : h;
    const float* p = O + ((long long)b * 2 * C + 2 * c) * L + t;
    float v = p[0] * sc[2 * c] + sh[2 * c] + p[L] * sc[2 * c + 1] + sh[2 * c + 1];
    Y[idx] = v;
}

// Nearest resize (ri = h*ih/oh) with optional residual add.
__global__ __launch_bounds__(256)
void k_resize_add(const float* __restrict__ Xs, const float* __restrict__ R,
                  float* __restrict__ Y, int C, int ih, int iw, int oh, int ow)
{
    long long total = (long long)C * oh * ow;
    long long idx = blockIdx.x * 256ll + threadIdx.x;
    if (idx >= total) return;
    int w = (int)(idx % ow);
    int h = (int)((idx / ow) % oh);
    int c = (int)(idx / ((long long)ow * oh));
    int ri = (int)(((long long)h * ih) / oh);
    int ci = (int)(((long long)w * iw) / ow);
    float v = Xs[((long long)c * ih + ri) * iw + ci];
    if (R) v += R[idx];
    Y[idx] = v;
}

// ---------------------------------------------------------------------------
// Host orchestration.
// ---------------------------------------------------------------------------
namespace {

struct Ws {
    char*  base;
    size_t cap;
    size_t off;
    float* alloc(size_t elems) {
        size_t bytes = (elems * sizeof(float) + 255) & ~(size_t)255;
        float* r = (float*)(base + off);
        off += bytes;
        return r;
    }
    size_t mark() const { return off; }
    void reset(size_t m) { off = m; }
};

// cast weights fp32 -> f16 into scratch, then run the WMMA conv.
// Scratch is released after the launch (stream-ordered reuse is safe).
inline void conv_launch(const float* X, const float* Wf, const float* B, float* Y,
                        int Ci, int Hin, int Win, int Co, int kh, int kw,
                        int st_, int pad, int batch, long long bsx, long long bsy,
                        Ws& ws, hipStream_t s)
{
    int Hout = (Hin + 2 * pad - kh) / st_ + 1;
    int Wout = (Win + 2 * pad - kw) / st_ + 1;
    int NSP  = Hout * Wout;
    long long nk = (long long)Co * Ci * kh * kw;
    size_t mk = ws.mark();
    _Float16* W16 = (_Float16*)ws.alloc((size_t)((nk + 1) / 2));
    k_cast_f16<<<(unsigned)((nk + 255) / 256), 256, 0, s>>>(Wf, W16, nk);
    dim3 g((NSP + BND - 1) / BND, (Co + BM - 1) / BM, batch);
    k_conv_wmma<<<g, 256, 0, s>>>(X, W16, B, Y, Ci, Hin, Win, Co, Hout, Wout,
                                  kh, kw, st_, pad, bsx, bsy);
    ws.reset(mk);
}

inline void elemwise(const float* X, const float* sc, const float* sh,
                     const float* res, float* Y, long long total, int C,
                     long long inner, int relu, hipStream_t s)
{
    k_bn_apply<<<(unsigned)((total + 255) / 256), 256, 0, s>>>(
        X, sc, sh, res, Y, total, C, inner, relu);
}

// conv -> BN(stats over batch) -> ReLU
inline void basic_block(const float* X, const float* w, const float* b,
                        const float* g, const float* be, float* Y,
                        int Ci, int Hin, int Co, int k, int st_, int pad,
                        Ws& ws, hipStream_t s)
{
    int Ho = (Hin + 2 * pad - k) / st_ + 1;
    size_t sp = (size_t)Ho * Ho;
    size_t mk = ws.mark();
    float* raw = ws.alloc((size_t)Co * sp);
    conv_launch(X, w, b, raw, Ci, Hin, Hin, Co, k, k, st_, pad, 1, 0, 0, ws, s);
    float* sc = ws.alloc(Co);
    float* sh = ws.alloc(Co);
    k_bn_stats<<<Co, 256, 0, s>>>(raw, g, be, sc, sh, (int)sp, 1, 0);
    elemwise(raw, sc, sh, nullptr, Y, (long long)Co * sp, Co, (long long)sp, 1, s);
    ws.reset(mk);
}

// One axial attention pass. Param offsets within axial_p (insertion order):
// +0 qkv_w, +1 bnq_g, +2 bnq_b, +3 bns_g, +4 bns_b, +5 bno_g, +6 bno_b, +7 rel
inline void axial_pass(const float* Xin, float* Yout, int C, int H, int W,
                       int width, int pb, void* const* din, Ws& ws, hipStream_t s)
{
    auto P = [&](int o) { return (const float*)din[pb + o]; };
    int L = width ? W : H;
    int B = width ? H : W;
    int gp = C / GROUPS;
    long long tot = (long long)C * H * W;

    float* T = ws.alloc((size_t)B * C * L);
    k_axial_gather<<<(unsigned)((tot + 255) / 256), 256, 0, s>>>(Xin, T, C, H, W, width);

    // qkv projection: batched GEMM (2C x C) x (C x L), batch = B
    float* QKV = ws.alloc((size_t)B * 2 * C * L);
    conv_launch(T, P(0), nullptr, QKV, C, 1, L, 2 * C, 1, 1, 1, 0,
                B, (long long)C * L, (long long)2 * C * L, ws, s);

    // bn1d over (b, t) per output channel, applied in place
    float* sc1 = ws.alloc(2 * C);
    float* sh1 = ws.alloc(2 * C);
    k_bn_stats<<<2 * C, 256, 0, s>>>(QKV, P(1), P(2), sc1, sh1, L, B,
                                     (long long)2 * C * L);
    long long tq = (long long)B * 2 * C * L;
    elemwise(QKV, sc1, sh1, nullptr, QKV, tq, 2 * C, (long long)L, 0, s);

    // bn2d stats of the stacked (qk|qr|kr) scores, recomputed on the fly
    float* sc2 = ws.alloc(32);
    float* sh2 = ws.alloc(32);
    k_score_stats<<<3 * GROUPS, 256, 0, s>>>(QKV, P(7), P(3), P(4), sc2, sh2,
                                             C, L, B, gp);

    // fused BN + 3-way sum + softmax
    float* SIM = ws.alloc((size_t)B * GROUPS * L * L);
    k_softmax_scores<<<(unsigned)(B * GROUPS * L), 128, 0, s>>>(
        QKV, P(7), sc2, sh2, SIM, C, L, gp);

    // sv / sve
    float* O = ws.alloc((size_t)B * 2 * C * L);
    long long to = (long long)B * GROUPS * gp * L;
    k_attn_apply<<<(unsigned)((to + 255) / 256), 256, 0, s>>>(
        SIM, QKV, P(7), O, C, L, gp, to);

    // bno (bn1d) + pair-sum + transpose back to (C,H,W)
    float* sc3 = ws.alloc(2 * C);
    float* sh3 = ws.alloc(2 * C);
    k_bn_stats<<<2 * C, 256, 0, s>>>(O, P(5), P(6), sc3, sh3, L, B,
                                     (long long)2 * C * L);
    k_axial_out<<<(unsigned)((tot + 255) / 256), 256, 0, s>>>(
        O, sc3, sh3, Yout, C, H, W, width);
}

// combined() level. Param offsets within comb_p (insertion order):
// +0 proj | c1: +1 w,+2 b,+3 g,+4 be | c2: +5..+8 |
// attn: +9 down_w, +10 bn1_g, +11 bn1_b, h:+12..+19, wb:+20..+27,
//       +28 up_w, +29 bn2_g, +30 bn2_b | out: +31 w,+32 b,+33 g,+34 be
inline float* run_combined(const float* xin, int l, void* const* din,
                           Ws& ws, hipStream_t s)
{
    static const int HIN[4] = {236, 118, 59, 29};
    static const int KSV[4] = {118, 59, 29, 14};
    int Ci = 64 << l, Co = 128 << l;
    int Hi = HIN[l], Ho = KSV[l];
    int base = 5 + 35 * l;
    auto P = [&](int o) { return (const float*)din[base + o]; };
    size_t sp = (size_t)Ho * Ho;

    float* out = ws.alloc((size_t)Co * sp);   // persists (returned)
    size_t mk = ws.mark();

    // projection conv: kernel s=2, stride 2, no pad, no bias
    float* xproj = ws.alloc((size_t)Co * sp);
    conv_launch(xin, P(0), nullptr, xproj, Ci, Hi, Hi, Co, 2, 2, 2, 0,
                1, 0, 0, ws, s);

    // CNN branch: two 3x3 basics
    float* t1 = ws.alloc((size_t)Co * sp);
    basic_block(xproj, P(1), P(2), P(3), P(4), t1, Co, Ho, Co, 3, 1, 1, ws, s);
    float* t2 = ws.alloc((size_t)Co * sp);
    basic_block(t1, P(5), P(6), P(7), P(8), t2, Co, Ho, Co, 3, 1, 1, ws, s);

    // Attention branch
    int Ic = Co / 2;
    float* araw = ws.alloc((size_t)Ic * sp);
    conv_launch(xproj, P(9), nullptr, araw, Co, Ho, Ho, Ic, 1, 1, 1, 0,
                1, 0, 0, ws, s);
    float* down = ws.alloc((size_t)Ic * sp);
    {
        float* sc = ws.alloc(Ic);
        float* sh = ws.alloc(Ic);
        k_bn_stats<<<Ic, 256, 0, s>>>(araw, P(10), P(11), sc, sh, (int)sp, 1, 0);
        elemwise(araw, sc, sh, nullptr, down, (long long)Ic * sp, Ic,
                 (long long)sp, 1, s);
    }
    float* ax1 = ws.alloc((size_t)Ic * sp);
    {
        size_t m2 = ws.mark();
        axial_pass(down, ax1, Ic, Ho, Ho, 0, base + 12, din, ws, s);
        ws.reset(m2);
    }
    float* ax2 = ws.alloc((size_t)Ic * sp);
    {
        size_t m2 = ws.mark();
        axial_pass(ax1, ax2, Ic, Ho, Ho, 1, base + 20, din, ws, s);
        ws.reset(m2);
    }
    // relu (in place)
    elemwise(ax2, nullptr, nullptr, nullptr, ax2, (long long)Ic * sp, Ic,
             (long long)sp, 1, s);
    // up conv 1x1 + bn2 + residual(xproj) + relu
    float* upraw = ws.alloc((size_t)Co * sp);
    conv_launch(ax2, P(28), nullptr, upraw, Ic, Ho, Ho, Co, 1, 1, 1, 0,
                1, 0, 0, ws, s);
    float* attn = ws.alloc((size_t)Co * sp);
    {
        float* sc = ws.alloc(Co);
        float* sh = ws.alloc(Co);
        k_bn_stats<<<Co, 256, 0, s>>>(upraw, P(29), P(30), sc, sh, (int)sp, 1, 0);
        elemwise(upraw, sc, sh, xproj, attn, (long long)Co * sp, Co,
                 (long long)sp, 1, s);
    }
    // (cnn + attn) -> out basic (1x1)
    float* sum = ws.alloc((size_t)Co * sp);
    elemwise(t2, nullptr, nullptr, attn, sum, (long long)Co * sp, Co,
             (long long)sp, 0, s);
    basic_block(sum, P(31), P(32), P(33), P(34), out, Co, Ho, Co, 1, 1, 0, ws, s);

    ws.reset(mk);
    return out;
}

} // namespace

// ---------------------------------------------------------------------------
// Input index map (setup_inputs insertion order):
// 0:x | stem 1:w 2:b 3:g 4:be | l1..l4: base 5+35*l (35 tensors each, see
// run_combined) | 145:dec1_w 146:dec1_b 147:dec2_w 148:dec2_b 149:cls_w 150:cls_b
// ---------------------------------------------------------------------------
extern "C" void kernel_launch(void* const* d_in, const int* in_sizes, int n_in,
                              void* d_out, int out_size, void* d_ws, size_t ws_size,
                              hipStream_t stream)
{
    (void)in_sizes; (void)n_in; (void)out_size;
    auto P = [&](int i) { return (const float*)d_in[i]; };
    const float* x = P(0);
    Ws ws{(char*)d_ws, ws_size, 0};

    // stem: 3 -> 64, 3x3 s2 p1, 472 -> 236   (kept for the decoder skip)
    float* low = ws.alloc(64ull * 236 * 236);
    basic_block(x, P(1), P(2), P(3), P(4), low, 3, 472, 64, 3, 2, 1, ws, stream);

    float* l1 = run_combined(low, 0, d_in, ws, stream);
    float* l2 = run_combined(l1, 1, d_in, ws, stream);   // kept for skip
    float* l3 = run_combined(l2, 2, d_in, ws, stream);
    float* l4 = run_combined(l3, 3, d_in, ws, stream);

    // Decoder
    size_t mk = ws.mark();
    float* d1 = ws.alloc(256ull * 14 * 14);
    conv_launch(l4, P(145), P(146), d1, 1024, 14, 14, 256, 1, 1, 1, 0,
                1, 0, 0, ws, stream);
    float* r1 = ws.alloc(256ull * 59 * 59);
    {
        long long t = 256ll * 59 * 59;
        k_resize_add<<<(unsigned)((t + 255) / 256), 256, 0, stream>>>(
            d1, l2, r1, 256, 14, 14, 59, 59);
    }
    float* d2 = ws.alloc(64ull * 59 * 59);
    conv_launch(r1, P(147), P(148), d2, 256, 59, 59, 64, 1, 1, 1, 0,
                1, 0, 0, ws, stream);
    float* r2 = ws.alloc(64ull * 236 * 236);
    {
        long long t = 64ll * 236 * 236;
        k_resize_add<<<(unsigned)((t + 255) / 256), 256, 0, stream>>>(
            d2, low, r2, 64, 59, 59, 236, 236);
    }
    float* r3 = ws.alloc(64ull * 472 * 472);
    {
        long long t = 64ll * 472 * 472;
        k_resize_add<<<(unsigned)((t + 255) / 256), 256, 0, stream>>>(
            r2, nullptr, r3, 64, 236, 236, 472, 472);
    }
    // classifier 64 -> 21 at 472x472, straight into d_out
    conv_launch(r3, P(149), P(150), (float*)d_out, 64, 472, 472, 21,
                1, 1, 1, 0, 1, 0, 0, ws, stream);
    ws.reset(mk);
}
